// VectorQuantizer_8899172237624
// MI455X (gfx1250) — matrix operand order repaired and checked by reading, not compiled, not run on hardware
//
#include <hip/hip_runtime.h>
#include <hip/hip_bf16.h>

// CDNA5 / gfx1250, wave32. WMMA bf16 16x16x32 path.
typedef __attribute__((ext_vector_type(16))) __bf16 v16bf;
typedef __attribute__((ext_vector_type(8)))  float  v8f;

#define VQ_K     1024      // codebook entries
#define VQ_D     256       // embedding dim
#define VQ_HW    4096      // 64*64 spatial
#define VQ_N     65536     // 16*4096 rows
#define VQ_BETA  0.25f

// ---------------------------------------------------------------------------
// Prep A: split codebook into bf16 hi/lo, pre-transposed + k-tiled:
//   Ehi/Elo layout: [kt=0..63][c=0..255][n=0..15]  (contiguous 8KB per k-tile)
// ---------------------------------------------------------------------------
__global__ void vq_prep_split(const float* __restrict__ emb,
                              __bf16* __restrict__ Ehi,
                              __bf16* __restrict__ Elo) {
    int gid = blockIdx.x * 256 + threadIdx.x;       // 0 .. 1024*256-1
    int k = gid >> 8;
    int c = gid & 255;
    float v = emb[gid];                             // emb[k*256 + c]
    __bf16 h = (__bf16)v;
    __bf16 l = (__bf16)(v - (float)h);
    int kt = k >> 4, n = k & 15;
    int dst = (kt * 256 + c) * 16 + n;
    Ehi[dst] = h;
    Elo[dst] = l;
}

// Prep B: per-code squared norms ||e_k||^2
__global__ void vq_prep_norm(const float* __restrict__ emb,
                             float* __restrict__ enorm) {
    int k = blockIdx.x * 256 + threadIdx.x;         // grid = 4 blocks
    const float* row = emb + k * VQ_D;
    float s = 0.0f;
    #pragma unroll 8
    for (int c = 0; c < VQ_D; ++c) { float v = row[c]; s += v * v; }
    enorm[k] = s;
}

// ---------------------------------------------------------------------------
// Main: 512 blocks x 256 threads (8 waves). Each wave = one 16-row tile.
// Split-bf16 matmul (xh*eh + xh*el + xl*eh) via v_wmma_f32_16x16x32_bf16.
// Double-buffered LDS B-stage: one barrier per k-tile, global L2 loads of the
// next tile overlap the 24 WMMAs of the current tile.
// ---------------------------------------------------------------------------
__launch_bounds__(256)
__global__ void vq_main(const float* __restrict__ latents,
                        const float* __restrict__ emb,
                        const __bf16* __restrict__ Ehi,
                        const __bf16* __restrict__ Elo,
                        const float* __restrict__ enorm,
                        float* __restrict__ out,
                        float* __restrict__ partial) {
    __shared__ __align__(16) __bf16 sBh[2][4096];  // 256 ch x 16 codes (hi)
    __shared__ __align__(16) __bf16 sBl[2][4096];  // 256 ch x 16 codes (lo)
    __shared__ int   sIdx[8][16];
    __shared__ float sRed[256];

    const int t    = threadIdx.x;
    const int wave = t >> 5;
    const int lane = t & 31;
    const int M    = lane & 15;        // row within tile
    const int half = lane >> 4;        // 0: lanes 0-15, 1: lanes 16-31

    const int  tile  = blockIdx.x * 8 + wave;
    const int  n0    = tile << 4;                  // first flat row
    const int  b     = n0 >> 12;                   // batch
    const int  hw    = n0 & 4095;                  // spatial base
    const long bbase = (long)b << 20;              // b * 256 * 4096

    // ---- Build A fragments: 16x32 bf16 per depth step, hi + lo ----------
    // ISA 16-bit A layout: lanes 0-15 -> M=0..15, VGPR j holds K pairs
    //   lanes<16 : j0..3 -> K{0,1},{2,3},{4,5},{6,7}; j4..7 -> K16..23
    //   lanes>=16: j0..3 -> K8..15;                   j4..7 -> K24..31
    const float* xrow = latents + bbase + hw + M;  // + c*VQ_HW per channel
    v16bf Ah[8], Al[8];
    #pragma unroll
    for (int d = 0; d < 8; ++d) {
        #pragma unroll
        for (int j = 0; j < 8; ++j) {
            int cj = (j < 4 ? 2 * j : 16 + 2 * (j - 4)) + half * 8;
            int c  = d * 32 + cj;
            float x0 = xrow[(long)c * VQ_HW];
            float x1 = xrow[(long)(c + 1) * VQ_HW];
            __bf16 h0 = (__bf16)x0, h1 = (__bf16)x1;
            Ah[d][2 * j]     = h0;
            Ah[d][2 * j + 1] = h1;
            Al[d][2 * j]     = (__bf16)(x0 - (float)h0);
            Al[d][2 * j + 1] = (__bf16)(x1 - (float)h1);
        }
    }

    float minv[8];
    int   mini[8];
    #pragma unroll
    for (int r = 0; r < 8; ++r) { minv[r] = 3.0e38f; mini[r] = 0; }

    // ---- K loop: 64 tiles of 16 codes, double-buffered B stage ----------
    const uint4* gEh = (const uint4*)Ehi;   // 512 uint4 per k-tile (8KB)
    const uint4* gEl = (const uint4*)Elo;

    // prologue: stage k-tile 0 into buffer 0
    uint4 ph0 = gEh[t];
    uint4 ph1 = gEh[t + 256];
    uint4 pl0 = gEl[t];
    uint4 pl1 = gEl[t + 256];
    {
        uint4* sH = (uint4*)sBh[0];
        uint4* sL = (uint4*)sBl[0];
        sH[t] = ph0; sH[t + 256] = ph1;
        sL[t] = pl0; sL[t + 256] = pl1;
    }
    __syncthreads();

    for (int kt = 0; kt < 64; ++kt) {
        const int cur = kt & 1;
        const int nxt = kt + 1;

        // issue next tile's L2 loads now; consumed only at the store below
        if (nxt < 64) {
            ph0 = gEh[nxt * 512 + t];
            ph1 = gEh[nxt * 512 + t + 256];
            pl0 = gEl[nxt * 512 + t];
            pl1 = gEl[nxt * 512 + t + 256];
        }
        // warm GL2 two tiles ahead (global_prefetch_b8)
        if (kt + 2 < 64) {
            __builtin_prefetch(gEh + (kt + 2) * 512 + t, 0, 1);
            __builtin_prefetch(gEl + (kt + 2) * 512 + t, 0, 1);
        }

        const int   code = (kt << 4) + M;
        const float en   = enorm[code];   // hoisted: hides under WMMAs

        v8f acc0 = {};   // xh*eh
        v8f acc1 = {};   // xh*el
        v8f acc2 = {};   // xl*eh
        const __bf16* bhbase = sBh[cur];
        const __bf16* blbase = sBl[cur];
        #pragma unroll
        for (int d = 0; d < 8; ++d) {
            // B layout: lane%16 = code, element s -> channel d*32 + k(half,s),
            // same K order as A (k = 8*half+s for s<8, 16+8*half+(s-8) for s>=8).
            v16bf bh, bl;
            #pragma unroll
            for (int s = 0; s < 16; ++s) {
                int kk = (s < 8 ? s : 8 + s) + half * 8;
                bh[s] = bhbase[(d * 32 + kk) * 16 + M];
                bl[s] = blbase[(d * 32 + kk) * 16 + M];
            }
            acc0 = __builtin_amdgcn_wmma_f32_16x16x32_bf16(
                       false, Ah[d], false, bh, (short)0, acc0, false, false);
            acc1 = __builtin_amdgcn_wmma_f32_16x16x32_bf16(
                       false, Ah[d], false, bl, (short)0, acc1, false, false);
            acc2 = __builtin_amdgcn_wmma_f32_16x16x32_bf16(
                       false, Al[d], false, bh, (short)0, acc2, false, false);
        }

        // D layout: VGPR r -> row r (lanes 0-15) / row r+8 (lanes 16-31), N = lane%16
        #pragma unroll
        for (int r = 0; r < 8; ++r) {
            float dot = acc0[r] + (acc1[r] + acc2[r]);
            float s   = en - 2.0f * dot;   // ||x||^2 constant per row: drop it
            if (s < minv[r]) { minv[r] = s; mini[r] = code; }
        }

        // store prefetched tile into the other buffer; its last readers
        // finished before the barrier that ended iteration kt-1.
        if (nxt < 64) {
            uint4* sH = (uint4*)sBh[nxt & 1];
            uint4* sL = (uint4*)sBl[nxt & 1];
            sH[t] = ph0; sH[t + 256] = ph1;
            sL[t] = pl0; sL[t + 256] = pl1;
        }
        __syncthreads();   // single barrier per k-tile
    }

    // ---- argmin across the 16 lanes of each half (first-min tiebreak) ---
    #pragma unroll
    for (int m = 1; m < 16; m <<= 1) {
        #pragma unroll
        for (int r = 0; r < 8; ++r) {
            float ov = __shfl_xor(minv[r], m, 32);
            int   oi = __shfl_xor(mini[r], m, 32);
            if (ov < minv[r] || (ov == minv[r] && oi < mini[r])) {
                minv[r] = ov; mini[r] = oi;
            }
        }
    }
    if (lane == 0) {
        #pragma unroll
        for (int r = 0; r < 8; ++r) sIdx[wave][r] = mini[r];
    }
    if (lane == 16) {
        #pragma unroll
        for (int r = 0; r < 8; ++r) sIdx[wave][8 + r] = mini[r];
    }
    __syncthreads();

    // ---- gather q, transposed store, loss accumulation ------------------
    const int    code = sIdx[wave][M];
    const float* qrow = emb + code * VQ_D;
    float lacc = 0.0f;
    #pragma unroll 4
    for (int it = 0; it < 128; ++it) {
        int  c = it * 2 + half;
        long a = bbase + (long)c * VQ_HW + hw + M;
        float q  = qrow[c];          // L1/L2-resident gather
        float x  = latents[a];
        float dd = q - x;
        lacc += dd * dd;
        out[a] = q;                  // q_st == q numerically
    }

    // deterministic block reduction of loss
    sRed[t] = lacc;
    __syncthreads();
    #pragma unroll
    for (int off = 128; off > 0; off >>= 1) {
        if (t < off) sRed[t] += sRed[t + off];
        __syncthreads();
    }
    if (t == 0) partial[blockIdx.x] = sRed[0];
}

// ---------------------------------------------------------------------------
// Finalize: deterministic reduction of 512 partials -> vq_loss scalar
// ---------------------------------------------------------------------------
__global__ void vq_finalize(const float* __restrict__ partial,
                            float* __restrict__ loss_out) {
    __shared__ float s[256];
    int t = threadIdx.x;
    s[t] = partial[t] + partial[t + 256];
    __syncthreads();
    #pragma unroll
    for (int off = 128; off > 0; off >>= 1) {
        if (t < off) s[t] += s[t + off];
        __syncthreads();
    }
    if (t == 0)
        loss_out[0] = s[0] * (1.0f + VQ_BETA) / 16777216.0f; // 1.25 * mean
}

extern "C" void kernel_launch(void* const* d_in, const int* in_sizes, int n_in,
                              void* d_out, int out_size, void* d_ws, size_t ws_size,
                              hipStream_t stream) {
    const float* latents = (const float*)d_in[0];   // [16,256,64,64] fp32
    const float* emb     = (const float*)d_in[1];   // [1024,256] fp32
    float*       out     = (float*)d_out;           // 16777216 q + 1 loss

    char*   ws      = (char*)d_ws;
    float*  enorm   = (float*)ws;                              // 4 KB
    __bf16* Ehi     = (__bf16*)(ws + 4096);                    // 512 KB
    __bf16* Elo     = (__bf16*)(ws + 4096 + 524288);           // 512 KB
    float*  partial = (float*)(ws + 4096 + 2 * 524288);        // 2 KB

    vq_prep_split<<<1024, 256, 0, stream>>>(emb, Ehi, Elo);
    vq_prep_norm <<<4,    256, 0, stream>>>(emb, enorm);
    vq_main      <<<512,  256, 0, stream>>>(latents, emb, Ehi, Elo, enorm,
                                            out, partial);
    vq_finalize  <<<1,    256, 0, stream>>>(partial, out + 16777216);
}
